// MultiHeadAttention_18537078850218
// MI455X (gfx1250) — compile-verified, matrix operand order checked by reading, not run on hardware
//
#include <hip/hip_runtime.h>

#define D_MODEL 1024
#define NH 16
#define DK 64
#define BB 2
#define SS 2048
#define SCALE 0.125f   // 1/sqrt(64)

typedef __attribute__((ext_vector_type(16))) __bf16 bf16x16;
typedef __attribute__((ext_vector_type(8)))  float  f32x8;
typedef unsigned short u16;
typedef unsigned int   u32;

union FragU { uint4 u[2]; bf16x16 v; };

__device__ __forceinline__ u16 f2bf(float x) {
  u32 u = __builtin_bit_cast(u32, x);
  u += 0x7FFFu + ((u >> 16) & 1u);   // round-to-nearest-even
  return (u16)(u >> 16);
}

__device__ __forceinline__ u32 pk2(float a, float b) {
  return (u32)f2bf(a) | ((u32)f2bf(b) << 16);
}

__device__ __forceinline__ uint4 pk8(f32x8 v, float s) {
  uint4 r;
  r.x = pk2(v[0] * s, v[1] * s);
  r.y = pk2(v[2] * s, v[3] * s);
  r.z = pk2(v[4] * s, v[5] * s);
  r.w = pk2(v[6] * s, v[7] * s);
  return r;
}

// Unified A/B fragment loader (shared per-lane quartile layout):
// lane l -> row (row + (l&15)), 8 contiguous bf16 at k + (l>>4)*8 and +16.
__device__ __forceinline__ bf16x16 load_frag(const u16* base, int row, int stride, int k) {
  int l = threadIdx.x & 31;
  const u16* p = base + (size_t)(row + (l & 15)) * stride + k + ((l >> 4) << 3);
  FragU f;
  f.u[0] = *(const uint4*)p;
  f.u[1] = *(const uint4*)(p + 16);
  return f.v;
}

__device__ __forceinline__ f32x8 wmma_bf16(bf16x16 a, bf16x16 b, f32x8 c) {
  return __builtin_amdgcn_wmma_f32_16x16x32_bf16(false, a, false, b, (short)0, c, false, false);
}

// ---- CDNA5 async global->LDS copy (16B per lane), tracked by ASYNCcnt ----
__device__ __forceinline__ void async_cp16(const u16* lds_dst, const u16* gsrc) {
  unsigned loff = (unsigned)(size_t)lds_dst;            // low 32 bits = LDS byte address
  unsigned long long ga = (unsigned long long)(size_t)gsrc;
  asm volatile("global_load_async_to_lds_b128 %0, %1, off"
               :: "v"(loff), "v"(ga) : "memory");
}

__device__ __forceinline__ void wait_async0() {
#if __has_builtin(__builtin_amdgcn_s_wait_asynccnt)
  __builtin_amdgcn_s_wait_asynccnt(0);
#else
  asm volatile("s_wait_asynccnt 0x0" ::: "memory");
#endif
}

// ---------- elementwise fp32 -> bf16 ----------
__global__ void cvt_kernel(const float* __restrict__ src, u16* __restrict__ dst, int n) {
  int i = (blockIdx.x * blockDim.x + threadIdx.x) * 8;
  if (i >= n) return;
  float4 a = *(const float4*)(src + i);
  float4 c = *(const float4*)(src + i + 4);
  uint4 o;
  o.x = pk2(a.x, a.y); o.y = pk2(a.z, a.w);
  o.z = pk2(c.x, c.y); o.w = pk2(c.z, c.w);
  *(uint4*)(dst + i) = o;
}

// W [H][D][DK] fp32 -> Wt [H][DK][D] bf16 (per-head transpose)
__global__ void wtr_kernel(const float* __restrict__ src, u16* __restrict__ dst) {
  int i = blockIdx.x * blockDim.x + threadIdx.x;
  if (i >= NH * D_MODEL * DK) return;
  int h  = i >> 16;
  int r  = i & 65535;
  int d  = r >> 6;
  int kk = r & 63;
  dst[((size_t)h * DK + kk) * D_MODEL + d] = f2bf(src[i]);
}

// ---------- async staging of a 64-row x 128-k weight tile into LDS ----------
#define WSTR 136   // LDS row stride (u16): 272B, 16B-aligned, conflict-free halves

__device__ __forceinline__ void stage_w128(const u16* W, int kt, u16* dst) {
  int tid = threadIdx.x;                       // 128 threads
#pragma unroll
  for (int j = 0; j < 8; ++j) {
    int c   = tid + 128 * j;                   // 0..1023 chunks of 16B
    int row = c >> 4;                          // 0..63
    int ch  = (c & 15) * 8;                    // 0..120
    async_cp16(dst + row * WSTR + ch, W + (size_t)row * D_MODEL + kt + ch);
  }
}

// ---------- QKV projection (wave tile: 32 s x 64 dk) ----------
// z==0/1 (Q/K): D = Wt[h] * X^T  -> store qh/kh [b,h,s,dk]
// z==2   (V) : D = X * Wt[h]-cols -> store vT [b,h,dk,s]
__global__ void __launch_bounds__(128) proj_kernel(
    const u16* __restrict__ qb, const u16* __restrict__ kb, const u16* __restrict__ vbx,
    const u16* __restrict__ Wqt, const u16* __restrict__ Wkt, const u16* __restrict__ Wvt,
    u16* __restrict__ qh, u16* __restrict__ kh, u16* __restrict__ vT) {
  __shared__ u16 wl[2][64 * WSTR];
  int z = blockIdx.z, h = blockIdx.y;
  int w  = threadIdx.x >> 5;
  int l  = threadIdx.x & 31, lr = l & 15, lh = l >> 4;
  int m0 = blockIdx.x * 128;               // flattened row base over B*S
  int b  = m0 / SS, s0 = m0 % SS;
  const u16* X = (z == 0) ? qb : (z == 1) ? kb : vbx;
  const u16* W = ((z == 0) ? Wqt : (z == 1) ? Wkt : Wvt) + (size_t)h * DK * D_MODEL;
  int sw = m0 + w * 32;                    // this wave's 32 rows (flattened)

  stage_w128(W, 0, &wl[0][0]);
  wait_async0();
  __syncthreads();
  int cur = 0;

  if (z < 2) {
    f32x8 acc[4][2] = {};
    bf16x16 xb0 = load_frag(X, sw, D_MODEL, 0);
    bf16x16 xb1 = load_frag(X, sw + 16, D_MODEL, 0);
    for (int kt = 0; kt < D_MODEL; kt += 128) {
      if (kt + 128 < D_MODEL) stage_w128(W, kt + 128, &wl[cur ^ 1][0]);
      const u16* wlc = &wl[cur][0];
#pragma unroll
      for (int j = 0; j < 4; ++j) {
        int kn = (kt + 32 * (j + 1)) & (D_MODEL - 1);   // register prefetch
        bf16x16 nx0 = load_frag(X, sw, D_MODEL, kn);
        bf16x16 nx1 = load_frag(X, sw + 16, D_MODEL, kn);
#pragma unroll
        for (int t = 0; t < 4; ++t) {
          bf16x16 a = load_frag(wlc, 16 * t, WSTR, 32 * j);
          acc[t][0] = wmma_bf16(a, xb0, acc[t][0]);
          acc[t][1] = wmma_bf16(a, xb1, acc[t][1]);
        }
        xb0 = nx0; xb1 = nx1;
      }
      wait_async0();
      __syncthreads();
      cur ^= 1;
    }
    u16* outp = (z == 0) ? qh : kh;
#pragma unroll
    for (int j = 0; j < 2; ++j) {
      u16* row = outp + ((size_t)(b * NH + h) * SS + (s0 + w * 32 + j * 16 + lr)) * DK;
#pragma unroll
      for (int t = 0; t < 4; ++t)
        *(uint4*)(row + 16 * t + 8 * lh) = pk8(acc[t][j], 1.f);
    }
  } else {
    f32x8 acc[2][4] = {};
    bf16x16 xa0 = load_frag(X, sw, D_MODEL, 0);
    bf16x16 xa1 = load_frag(X, sw + 16, D_MODEL, 0);
    for (int kt = 0; kt < D_MODEL; kt += 128) {
      if (kt + 128 < D_MODEL) stage_w128(W, kt + 128, &wl[cur ^ 1][0]);
      const u16* wlc = &wl[cur][0];
#pragma unroll
      for (int j = 0; j < 4; ++j) {
        int kn = (kt + 32 * (j + 1)) & (D_MODEL - 1);
        bf16x16 nx0 = load_frag(X, sw, D_MODEL, kn);
        bf16x16 nx1 = load_frag(X, sw + 16, D_MODEL, kn);
#pragma unroll
        for (int t = 0; t < 4; ++t) {
          bf16x16 wb = load_frag(wlc, 16 * t, WSTR, 32 * j);
          acc[0][t] = wmma_bf16(xa0, wb, acc[0][t]);
          acc[1][t] = wmma_bf16(xa1, wb, acc[1][t]);
        }
        xa0 = nx0; xa1 = nx1;
      }
      wait_async0();
      __syncthreads();
      cur ^= 1;
    }
#pragma unroll
    for (int j = 0; j < 2; ++j)
#pragma unroll
      for (int t = 0; t < 4; ++t) {
        u16* p = vT + ((size_t)(b * NH + h) * DK + 16 * t + lr) * SS
                    + (s0 + w * 32 + j * 16 + 8 * lh);
        *(uint4*)p = pk8(acc[j][t], 1.f);
      }
  }
}

// ---------- flash attention with async LDS staging of K/V tiles ----------
#define LPAD 72   // LDS row stride (u16) for 64-wide tiles, padded for banks

__device__ __forceinline__ void stage_tiles(const u16* kp, const u16* vp, int s0,
                                            u16* kdst, u16* vdst) {
  int tid = threadIdx.x;                 // 256 threads
#pragma unroll
  for (int j = 0; j < 2; ++j) {
    int c   = tid + 256 * j;             // 0..511 chunks of 16B
    int row = c >> 3;                    // 0..63
    int ch  = (c & 7) * 8;               // element offset in row
    async_cp16(kdst + row * LPAD + ch, kp + (size_t)(s0 + row) * DK + ch);
    async_cp16(vdst + row * LPAD + ch, vp + (size_t)row * SS + s0 + ch);
  }
}

__global__ void __launch_bounds__(256) attn_kernel(
    const u16* __restrict__ qh, const u16* __restrict__ kh,
    const u16* __restrict__ vT, u16* __restrict__ ctx) {
  __shared__ u16 kt[2][64 * LPAD];
  __shared__ u16 vt[2][64 * LPAD];
  int w = threadIdx.x >> 5;
  int l = threadIdx.x & 31, lr = l & 15, lh = l >> 4;
  int h = blockIdx.y, b = blockIdx.z;
  int q0 = blockIdx.x * 128 + w * 16;
  const u16* qp = qh + (size_t)(b * NH + h) * SS * DK;   // [s][dk]
  const u16* kp = kh + (size_t)(b * NH + h) * SS * DK;   // [s][dk]
  const u16* vp = vT + (size_t)(b * NH + h) * DK * SS;   // [dk][s]

  bf16x16 qf0 = load_frag(qp, q0, DK, 0);    // Q^T B-frags (loop-invariant)
  bf16x16 qf1 = load_frag(qp, q0, DK, 32);

  f32x8 acc[4] = {};                          // ctx^T: M=dk (4 tiles), N=query
  float m_run = -3.0e38f, l_run = 0.f;

  stage_tiles(kp, vp, 0, &kt[0][0], &vt[0][0]);
  wait_async0();
  __syncthreads();

  int cur = 0;
  for (int s0 = 0; s0 < SS; s0 += 64) {
    if (s0 + 64 < SS)
      stage_tiles(kp, vp, s0 + 64, &kt[cur ^ 1][0], &vt[cur ^ 1][0]);
    const u16* kc = &kt[cur][0];
    const u16* vc = &vt[cur][0];

    f32x8 sc[4];                              // scores^T: M=key tiles, N=query
#pragma unroll
    for (int t = 0; t < 4; ++t) {
      f32x8 c = {};
      c = wmma_bf16(load_frag(kc, 16 * t, LPAD, 0), qf0, c);
      c = wmma_bf16(load_frag(kc, 16 * t, LPAD, 32), qf1, c);
      sc[t] = c;
    }
    // per-lane = per-query online softmax (merge lane halves via shfl_xor 16)
    float mx = m_run;
#pragma unroll
    for (int t = 0; t < 4; ++t)
#pragma unroll
      for (int i = 0; i < 8; ++i) { sc[t][i] *= SCALE; mx = fmaxf(mx, sc[t][i]); }
    mx = fmaxf(mx, __shfl_xor(mx, 16));
    float sum = 0.f;
#pragma unroll
    for (int t = 0; t < 4; ++t)
#pragma unroll
      for (int i = 0; i < 8; ++i) { float e = __expf(sc[t][i] - mx); sc[t][i] = e; sum += e; }
    sum += __shfl_xor(sum, 16);
    float corr = __expf(m_run - mx);
    l_run = l_run * corr + sum;
    m_run = mx;
#pragma unroll
    for (int t = 0; t < 4; ++t)
#pragma unroll
      for (int i = 0; i < 8; ++i) acc[t][i] *= corr;
    // exp'd scores^T fragments are already in B-frag layout for P^T
    FragU p0, p1;
    p0.u[0] = pk8(sc[0], 1.f); p0.u[1] = pk8(sc[1], 1.f);   // keys s0..s0+31
    p1.u[0] = pk8(sc[2], 1.f); p1.u[1] = pk8(sc[3], 1.f);   // keys s0+32..s0+63
#pragma unroll
    for (int t = 0; t < 4; ++t) {
      acc[t] = wmma_bf16(load_frag(vc, 16 * t, LPAD, 0), p0.v, acc[t]);
      acc[t] = wmma_bf16(load_frag(vc, 16 * t, LPAD, 32), p1.v, acc[t]);
    }
    wait_async0();        // next tile's async copies have landed
    __syncthreads();      // all waves done reading this buffer
    cur ^= 1;
  }
  float inv = 1.f / l_run;
  // ctx [b, s, h*64 + dk]: lane holds fixed query, 8 consecutive dk per frag
  u16* crow = ctx + ((size_t)b * SS + q0 + lr) * (NH * DK) + h * DK;
#pragma unroll
  for (int t = 0; t < 4; ++t)
    *(uint4*)(crow + 16 * t + 8 * lh) = pk8(acc[t], inv);
}

// ---------- output projection: out^T = W_o * ctx^T + b_o (wave: 32 s x 64 m) ----------
__global__ void __launch_bounds__(128) oproj_kernel(
    const u16* __restrict__ ctx, const u16* __restrict__ Wob,
    const float* __restrict__ b_o, float* __restrict__ out) {
  __shared__ u16 wl[2][64 * WSTR];
  int w = threadIdx.x >> 5;
  int l = threadIdx.x & 31, lr = l & 15, lh = l >> 4;
  int sw = blockIdx.x * 128 + w * 32;   // 32 s-columns for this wave
  int mb = blockIdx.y * 64;             // 64 output features
  const u16* W = Wob + (size_t)mb * D_MODEL;

  stage_w128(W, 0, &wl[0][0]);
  wait_async0();
  __syncthreads();
  int cur = 0;

  f32x8 acc[4][2] = {};
  bf16x16 cb0 = load_frag(ctx, sw, D_MODEL, 0);
  bf16x16 cb1 = load_frag(ctx, sw + 16, D_MODEL, 0);
  for (int kt = 0; kt < D_MODEL; kt += 128) {
    if (kt + 128 < D_MODEL) stage_w128(W, kt + 128, &wl[cur ^ 1][0]);
    const u16* wlc = &wl[cur][0];
#pragma unroll
    for (int j = 0; j < 4; ++j) {
      int kn = (kt + 32 * (j + 1)) & (D_MODEL - 1);
      bf16x16 nc0 = load_frag(ctx, sw, D_MODEL, kn);
      bf16x16 nc1 = load_frag(ctx, sw + 16, D_MODEL, kn);
#pragma unroll
      for (int t = 0; t < 4; ++t) {
        bf16x16 a = load_frag(wlc, 16 * t, WSTR, 32 * j);
        acc[t][0] = wmma_bf16(a, cb0, acc[t][0]);
        acc[t][1] = wmma_bf16(a, cb1, acc[t][1]);
      }
      cb0 = nc0; cb1 = nc1;
    }
    wait_async0();
    __syncthreads();
    cur ^= 1;
  }
#pragma unroll
  for (int j = 0; j < 2; ++j) {
    float* orow = out + (size_t)(sw + j * 16 + lr) * D_MODEL;
#pragma unroll
    for (int t = 0; t < 4; ++t) {
      int mm = mb + 16 * t + 8 * lh;
      float4 b0 = *(const float4*)(b_o + mm);
      float4 b1 = *(const float4*)(b_o + mm + 4);
      float4 o0 = make_float4(acc[t][j][0] + b0.x, acc[t][j][1] + b0.y,
                              acc[t][j][2] + b0.z, acc[t][j][3] + b0.w);
      float4 o1 = make_float4(acc[t][j][4] + b1.x, acc[t][j][5] + b1.y,
                              acc[t][j][6] + b1.z, acc[t][j][7] + b1.w);
      *(float4*)(orow + mm) = o0;
      *(float4*)(orow + mm + 4) = o1;
    }
  }
}

extern "C" void kernel_launch(void* const* d_in, const int* in_sizes, int n_in,
                              void* d_out, int out_size, void* d_ws, size_t ws_size,
                              hipStream_t stream) {
  const float* q   = (const float*)d_in[0];
  const float* k   = (const float*)d_in[1];
  const float* v   = (const float*)d_in[2];
  const float* W_q = (const float*)d_in[3];
  const float* W_k = (const float*)d_in[4];
  const float* W_v = (const float*)d_in[5];
  const float* W_o = (const float*)d_in[6];
  const float* b_o = (const float*)d_in[7];
  float* out = (float*)d_out;

  char* ws = (char*)d_ws;                 // 64 MiB total
  u16* qb  = (u16*)(ws + 0);              // [B,S,D] bf16
  u16* kb  = (u16*)(ws + 8388608);
  u16* vb  = (u16*)(ws + 16777216);
  u16* Wqt = (u16*)(ws + 25165824);       // [H,DK,D] bf16
  u16* Wkt = (u16*)(ws + 27262976);
  u16* Wvt = (u16*)(ws + 29360128);
  u16* Wob = (u16*)(ws + 31457280);       // [D,D] bf16
  u16* qh  = (u16*)(ws + 33554432);       // [B,H,S,DK] bf16
  u16* kh  = (u16*)(ws + 41943040);       // [B,H,S,DK] bf16
  u16* vT  = (u16*)(ws + 50331648);       // [B,H,DK,S] bf16
  u16* ctx = (u16*)(ws + 58720256);       // [B,S,H*DK] bf16

  const int nx = BB * SS * D_MODEL;       // 4194304
  const int nw = NH * D_MODEL * DK;       // 1048576
  cvt_kernel<<<nx / 8 / 256, 256, 0, stream>>>(q, qb, nx);
  cvt_kernel<<<nx / 8 / 256, 256, 0, stream>>>(k, kb, nx);
  cvt_kernel<<<nx / 8 / 256, 256, 0, stream>>>(v, vb, nx);
  cvt_kernel<<<nw / 8 / 256, 256, 0, stream>>>(W_o, Wob, nw);
  wtr_kernel<<<nw / 256, 256, 0, stream>>>(W_q, Wqt);
  wtr_kernel<<<nw / 256, 256, 0, stream>>>(W_k, Wkt);
  wtr_kernel<<<nw / 256, 256, 0, stream>>>(W_v, Wvt);

  proj_kernel<<<dim3(32, 16, 3), 128, 0, stream>>>(qb, kb, vb, Wqt, Wkt, Wvt, qh, kh, vT);
  attn_kernel<<<dim3(16, 16, 2), 256, 0, stream>>>(qh, kh, vT, ctx);
  oproj_kernel<<<dim3(32, 16), 128, 0, stream>>>(ctx, Wob, b_o, out);
}
